// UnidirAttention_25305947308627
// MI455X (gfx1250) — matrix operand order repaired and checked by reading, not compiled, not run on hardware
//
#include <hip/hip_runtime.h>
#include <hip/hip_bf16.h>

// ---- types ----
typedef __bf16 bf16_t;
typedef __attribute__((ext_vector_type(16))) __bf16 v16bf;
typedef __attribute__((ext_vector_type(8)))  __bf16 bf16x8;
typedef __attribute__((ext_vector_type(8)))  float  v8f;

union FragA { v16bf v; bf16x8 h[2]; };

// Problem constants (fixed by setup_inputs)
#define BB   2
#define SS   2048
#define HH   1024
#define H3   3072
#define NH   16
#define DD   64

// Cross-row exchange (lane <-> lane^16) in wave32 as a VALU op:
// v_permlanex16_b32 with identity lane selects. Replaces ds_bpermute-based
// __shfl_xor(v,16) on the softmax critical path.
__device__ __forceinline__ float permx16f(float v) {
  int i = __builtin_bit_cast(int, v);
  int r = __builtin_amdgcn_permlanex16(i, i, 0x76543210, 0xfedcba98, false, false);
  return __builtin_bit_cast(float, r);
}

// ---------------------------------------------------------------------------
// f32 -> bf16 convert
// ---------------------------------------------------------------------------
__global__ __launch_bounds__(256) void k_f32_to_bf16(const float* __restrict__ in,
                                                     bf16_t* __restrict__ out, int n) {
  int i = blockIdx.x * 256 + threadIdx.x;
  if (i < n) out[i] = (bf16_t)in[i];
}

// ---------------------------------------------------------------------------
// LDS-tiled bf16 WMMA GEMM with double-buffered ASYNC global->LDS DMA:
//   C[M,ldc] = A[M,K] * W[N,K]^T + bias[N]
// Block: 256 threads (8 waves). Tile: BM=128, BN=64, BK=32.
// Per K-step each thread issues 3x global_load_async_to_lds_b128 (ASYNCcnt);
// the next tile's DMA overlaps the current tile's WMMAs.
// ---------------------------------------------------------------------------
template <typename OutT>
__global__ __launch_bounds__(256)
void k_gemm_wmma(const bf16_t* __restrict__ A, const bf16_t* __restrict__ W,
                 const float* __restrict__ bias, OutT* __restrict__ C,
                 int M, int N, int K, int ldc) {
  const int LDA = 40, LDB = 40;               // padded LDS strides (16B aligned)
  __shared__ bf16_t As[2][128 * LDA];         // 2 x 10 KB
  __shared__ bf16_t Bs[2][64 * LDB];          // 2 x  5 KB

  const int tid  = threadIdx.x;
  const int lane = tid & 31;
  const int wv   = tid >> 5;
  const int wm   = wv & 3;                    // 0..3
  const int wn   = wv >> 2;                   // 0..1
  const int qc   = lane & 15;
  const int hi   = lane >> 4;                 // 0/1 (half-wave)
  const int m0   = blockIdx.y * 128;
  const int n0   = blockIdx.x * 64;

  // per-thread DMA slots: A units tid and tid+256 (rows 0..63 / 64..127), B unit tid
  const int ar = tid >> 2, ac = (tid & 3) * 8;

  auto issue = [&](int buf, int k0) {
    const bf16_t* ga0 = A + (size_t)(m0 + ar) * K + k0 + ac;
    const bf16_t* ga1 = A + (size_t)(m0 + 64 + ar) * K + k0 + ac;
    const bf16_t* gb  = W + (size_t)(n0 + ar) * K + k0 + ac;
    uint32_t la0 = (uint32_t)(uintptr_t)&As[buf][ar * LDA + ac];
    uint32_t la1 = (uint32_t)(uintptr_t)&As[buf][(64 + ar) * LDA + ac];
    uint32_t lb  = (uint32_t)(uintptr_t)&Bs[buf][ar * LDB + ac];
    asm volatile("global_load_async_to_lds_b128 %0, %1, off"
                 :: "v"(la0), "v"((uint64_t)(uintptr_t)ga0) : "memory");
    asm volatile("global_load_async_to_lds_b128 %0, %1, off"
                 :: "v"(la1), "v"((uint64_t)(uintptr_t)ga1) : "memory");
    asm volatile("global_load_async_to_lds_b128 %0, %1, off"
                 :: "v"(lb),  "v"((uint64_t)(uintptr_t)gb)  : "memory");
  };

  v8f acc[2][2] = {};
  const int nst = K / 32;
  issue(0, 0);

  for (int st = 0; st < nst; ++st) {
    const int cur = st & 1;
    if (st + 1 < nst) {
      issue(cur ^ 1, (st + 1) * 32);          // prefetch next tile (async DMA)
      asm volatile("s_wait_asynccnt 0x3" ::: "memory");  // current tile's 3 ops done
    } else {
      asm volatile("s_wait_asynccnt 0x0" ::: "memory");
    }
    __syncthreads();

    FragA af[2], bfr[2];
    for (int i = 0; i < 2; ++i) {             // A frags (16-bit A layout)
      int row = wm * 32 + i * 16 + qc;
      af[i].h[0] = *(const bf16x8*)&As[cur][row * LDA + hi * 8];
      af[i].h[1] = *(const bf16x8*)&As[cur][row * LDA + hi * 8 + 16];
    }
    for (int j = 0; j < 2; ++j) {             // B frags (same k order as A)
      int col = wn * 32 + j * 16 + qc;
      bfr[j].h[0] = *(const bf16x8*)&Bs[cur][col * LDB + hi * 8];
      bfr[j].h[1] = *(const bf16x8*)&Bs[cur][col * LDB + hi * 8 + 16];
    }
    for (int i = 0; i < 2; ++i)
      for (int j = 0; j < 2; ++j)
        acc[i][j] = __builtin_amdgcn_wmma_f32_16x16x32_bf16(
            false, af[i].v, false, bfr[j].v, (short)0, acc[i][j], false, false);
    __syncthreads();                          // all waves done reading buf[cur]
  }

  // epilogue: C layout lane=col, VGPR r = row r (+8 for high half)
  for (int j = 0; j < 2; ++j) {
    int n = n0 + wn * 32 + j * 16 + qc;
    float bv = bias[n];
    for (int i = 0; i < 2; ++i) {
      int mrow = m0 + wm * 32 + i * 16 + hi * 8;
      for (int r = 0; r < 8; ++r) {
        float v = acc[i][j][r] + bv;
        C[(size_t)(mrow + r) * ldc + n] = (OutT)v;
      }
    }
  }
}

// ---------------------------------------------------------------------------
// Repack V from qkv[B,S,3H] (cols 2048..3071) into Vt[B,h,d,S] (bf16)
// ---------------------------------------------------------------------------
__global__ __launch_bounds__(256) void k_make_vt(const bf16_t* __restrict__ qkv,
                                                 bf16_t* __restrict__ vt) {
  size_t i = (size_t)blockIdx.x * 256 + threadIdx.x;  // B*NH*DD*SS = 4194304
  int s = (int)(i & (SS - 1));
  size_t r = i >> 11;
  int dd = (int)(r & (DD - 1)); r >>= 6;
  int h = (int)(r & (NH - 1));
  int b = (int)(r >> 4);
  vt[i] = qkv[((size_t)(b * SS + s)) * H3 + 2 * HH + h * DD + dd];
}

// ---------------------------------------------------------------------------
// Flash attention, transposed formulation. One wave = 32 queries of one (b,h)
// (two 16-query subtiles sharing every K and V fragment -> 2x L2 reuse).
//   Lt = K(16k x 32d) x Qt(32d x 16q)     [WMMA per (key-subtile, d-step, qq)]
//   Ot += Vt(16dd x 32k) x Pt(32k x 16q)  [each V frag feeds both q-subtiles]
// Softmax stats are per-lane (column = query); all cross-lane traffic is
// v_permlanex16 (VALU), not LDS. Causal mask key>q -> -10000.
// ---------------------------------------------------------------------------
__global__ __launch_bounds__(256)
void k_attn(const bf16_t* __restrict__ qkv, const bf16_t* __restrict__ vt,
            bf16_t* __restrict__ aout) {
  const int lane = threadIdx.x & 31;
  const int wv   = threadIdx.x >> 5;
  const int h    = blockIdx.y;
  const int b    = blockIdx.z;
  const int q0   = (blockIdx.x * 8 + wv) * 32;
  const int qc   = lane & 15;
  const int hi   = lane >> 4;

  // Q^T B-fragments, pre-scaled by 1/sqrt(d)=0.125 (exact exponent shift)
  FragA qf[2][2];
  for (int qq = 0; qq < 2; ++qq) {
    const bf16_t* qrow = qkv + (size_t)(b * SS + q0 + qq * 16 + qc) * H3 + h * DD;
    for (int ks = 0; ks < 2; ++ks) {
      const bf16x8* p = (const bf16x8*)(qrow + ks * 32 + hi * 8);
      bf16x8 c0 = p[0], c1 = p[2];
      for (int e = 0; e < 8; ++e) {
        qf[qq][ks].h[0][e] = (__bf16)((float)c0[e] * 0.125f);
        qf[qq][ks].h[1][e] = (__bf16)((float)c1[e] * 0.125f);
      }
    }
  }

  v8f o[2][4] = {};
  float m[2] = {-3.0e38f, -3.0e38f}, l[2] = {0.0f, 0.0f};
  const bf16_t* kreg = qkv + (size_t)b * SS * H3 + HH + h * DD;    // K region
  const bf16_t* vreg = vt + ((size_t)(b * NH + h) * DD) * SS;      // Vt region
  const int nsteps = (q0 >> 5) + 1;                                // keys to q0+31

  for (int st = 0; st < nsteps; ++st) {
    const int kb = st * 32;
    __builtin_prefetch(kreg + (size_t)(kb + 32 + qc) * H3, 0, 1);

    // ---- logits^T: c[qq][t], K fragments shared by both q-subtiles ----
    v8f c[2][2];
    for (int t = 0; t < 2; ++t) {
      FragA kf[2];
      for (int ks = 0; ks < 2; ++ks) {
        const bf16_t* kr = kreg + (size_t)(kb + t * 16 + qc) * H3 + ks * 32 + hi * 8;
        kf[ks].h[0] = *(const bf16x8*)kr;
        kf[ks].h[1] = *(const bf16x8*)(kr + 16);
      }
      for (int qq = 0; qq < 2; ++qq) {
        v8f acc = {};
        for (int ks = 0; ks < 2; ++ks)
          acc = __builtin_amdgcn_wmma_f32_16x16x32_bf16(
              false, kf[ks].v, false, qf[qq][ks].v, (short)0, acc, false, false);
        c[qq][t] = acc;
      }
    }

    // ---- per q-subtile: mask, online softmax, build P^T B-fragment ----
    FragA pf[2];
    for (int qq = 0; qq < 2; ++qq) {
      const int q0q = q0 + qq * 16;
      if (kb + 31 > q0q) {                    // causal: masked value -10000
        const int q = q0q + qc;
        for (int t = 0; t < 2; ++t)
          for (int r = 0; r < 8; ++r) {
            int key = kb + t * 16 + hi * 8 + r;
            if (key > q) c[qq][t][r] = -10000.0f;
          }
      }
      float mb = c[qq][0][0];
      for (int r = 0; r < 8; ++r) {
        mb = fmaxf(mb, c[qq][0][r]);
        mb = fmaxf(mb, c[qq][1][r]);
      }
      mb = fmaxf(mb, permx16f(mb));           // combine the two key half-rows
      const float mnew  = fmaxf(m[qq], mb);
      const float scale = __expf(m[qq] - mnew);

      float ps0[8], ps1[8], ls = 0.0f;
      for (int r = 0; r < 8; ++r) {
        ps0[r] = __expf(c[qq][0][r] - mnew);
        ps1[r] = __expf(c[qq][1][r] - mnew);
        ls += ps0[r] + ps1[r];
      }
      ls += permx16f(ls);
      l[qq] = l[qq] * scale + ls;
      for (int d4 = 0; d4 < 4; ++d4)
        for (int e = 0; e < 8; ++e) o[qq][d4][e] *= scale;

      // C^T -> B-fragment: each half only needs the other half's ps0 or ps1,
      // so one permlanex16 per r carries both directions (8 exchanges total).
      for (int r = 0; r < 8; ++r) {
        float t_ = hi ? ps0[r] : ps1[r];      // value the other half needs
        float u  = permx16f(t_);              // low gets b0=ps0^16, high gets b1=ps1^16
        pf[qq].h[0][r] = (__bf16)(hi ? u : ps0[r]);
        pf[qq].h[1][r] = (__bf16)(hi ? ps1[r] : u);
      }
      m[qq] = mnew;
    }

    // ---- O^T += V^T x P^T, each V fragment feeds both q-subtiles ----
    for (int d4 = 0; d4 < 4; ++d4) {
      FragA vf;
      const bf16_t* vr = vreg + (size_t)(d4 * 16 + qc) * SS + kb + hi * 16;
      vf.h[0] = *(const bf16x8*)vr;
      vf.h[1] = *(const bf16x8*)(vr + 8);
      o[0][d4] = __builtin_amdgcn_wmma_f32_16x16x32_bf16(
          false, vf.v, false, pf[0].v, (short)0, o[0][d4], false, false);
      o[1][d4] = __builtin_amdgcn_wmma_f32_16x16x32_bf16(
          false, vf.v, false, pf[1].v, (short)0, o[1][d4], false, false);
    }
  }

  // ---- epilogue: normalize, store [B,S,h,d] bf16, 16B vector stores ----
  for (int qq = 0; qq < 2; ++qq) {
    const float rl = 1.0f / l[qq];
    bf16_t* orow = aout + (size_t)(b * SS + q0 + qq * 16 + qc) * HH + h * DD + hi * 8;
    for (int d4 = 0; d4 < 4; ++d4) {
      bf16x8 pk;
      for (int r = 0; r < 8; ++r) pk[r] = (__bf16)(o[qq][d4][r] * rl);
      *(bf16x8*)(orow + d4 * 16) = pk;
    }
  }
}

// ---------------------------------------------------------------------------
// launcher
// ---------------------------------------------------------------------------
extern "C" void kernel_launch(void* const* d_in, const int* in_sizes, int n_in,
                              void* d_out, int out_size, void* d_ws, size_t ws_size,
                              hipStream_t stream) {
  const float* x     = (const float*)d_in[0];
  const float* W_qkv = (const float*)d_in[1];
  const float* b_qkv = (const float*)d_in[2];
  const float* W_o   = (const float*)d_in[3];
  const float* b_o   = (const float*)d_in[4];
  float* out = (float*)d_out;

  char* ws = (char*)d_ws;
  bf16_t* xb    = (bf16_t*)(ws + 0);                  //  8 MB  [4096,1024]
  bf16_t* wqkvb = (bf16_t*)(ws + (8u << 20));         //  6 MB  [3072,1024]
  bf16_t* wob   = (bf16_t*)(ws + (14u << 20));        //  2 MB  [1024,1024]
  bf16_t* qkv   = (bf16_t*)(ws + (16u << 20));        // 24 MB  [4096,3072]
  bf16_t* vt    = (bf16_t*)(ws + (40u << 20));        //  8 MB  [2,16,64,2048]
  bf16_t* aout  = (bf16_t*)(ws + (48u << 20));        //  8 MB  [4096,1024]

  // 1) converts
  k_f32_to_bf16<<<(BB * SS * HH) / 256, 256, 0, stream>>>(x, xb, BB * SS * HH);
  k_f32_to_bf16<<<(3 * HH * HH) / 256, 256, 0, stream>>>(W_qkv, wqkvb, 3 * HH * HH);
  k_f32_to_bf16<<<(HH * HH) / 256, 256, 0, stream>>>(W_o, wob, HH * HH);

  // 2) QKV projection: [4096,1024] x [3072,1024]^T + b -> qkv bf16
  {
    dim3 grid(H3 / 64, (BB * SS) / 128);
    k_gemm_wmma<bf16_t><<<grid, 256, 0, stream>>>(xb, wqkvb, b_qkv, qkv,
                                                  BB * SS, H3, HH, H3);
  }

  // 3) V repack -> Vt[B,h,d,S]
  k_make_vt<<<(BB * NH * DD * SS) / 256, 256, 0, stream>>>(qkv, vt);

  // 4) causal attention -> aout bf16 [B,S,H]
  {
    dim3 grid(SS / (32 * 8), NH, BB);
    k_attn<<<grid, 256, 0, stream>>>(qkv, vt, aout);
  }

  // 5) output projection: [4096,1024] x [1024,1024]^T + b -> f32 out
  {
    dim3 grid(HH / 64, (BB * SS) / 128);
    k_gemm_wmma<float><<<grid, 256, 0, stream>>>(aout, wob, b_o, out,
                                                 BB * SS, HH, HH, HH);
  }
}
